// ComplexAttention_23536420782599
// MI455X (gfx1250) — matrix operand order repaired and checked by reading, not compiled, not run on hardware
//
#include <hip/hip_runtime.h>
#include <math.h>

#define DI __device__ __forceinline__

typedef __attribute__((ext_vector_type(16))) __bf16 v16bf;
typedef __attribute__((ext_vector_type(8)))  __bf16 v8bf;
typedef __attribute__((ext_vector_type(8)))  float  v8f;

static constexpr int B_  = 4;
static constexpr int S_  = 4096;
static constexpr int D_  = 256;
static constexpr int D2_ = 512;

// ---------- helpers ----------

DI __bf16 f2bf(float f) { return (__bf16)f; }   // native v_cvt on gfx1250

DI v8f zero8() {
  v8f z = {0.f, 0.f, 0.f, 0.f, 0.f, 0.f, 0.f, 0.f};
  return z;
}

// A-matrix 16x32 bf16 tile from an f32 row-major source (ld = elems).
// CDNA5 A layout: lane = M row (lane&15); lanes 0-15 hold K 0-7 / 16-23,
// lanes 16-31 hold K 8-15 / 24-31.
DI v16bf load_a_f32(const float* base, int ld, int lane) {
  const int m  = lane & 15;
  const int kb = (lane >> 4) * 8;
  const float* row = base + (size_t)m * ld + kb;
  v8f lo = *(const v8f*)(row);
  v8f hi = *(const v8f*)(row + 16);
  v8bf blo = __builtin_convertvector(lo, v8bf);   // v_cvt_pk_bf16_f32 x4
  v8bf bhi = __builtin_convertvector(hi, v8bf);
  return __builtin_shufflevector(blo, bhi, 0, 1, 2, 3, 4, 5, 6, 7,
                                 8, 9, 10, 11, 12, 13, 14, 15);
}

// A-matrix 16x32 tile from a bf16 row-major source.
DI v16bf load_a_bf16(const __bf16* base, int ld, int lane) {
  const int m  = lane & 15;
  const int kb = (lane >> 4) * 8;
  const __bf16* row = base + (size_t)m * ld + kb;
  v8bf lo = *(const v8bf*)(row);
  v8bf hi = *(const v8bf*)(row + 16);
  return __builtin_shufflevector(lo, hi, 0, 1, 2, 3, 4, 5, 6, 7,
                                 8, 9, 10, 11, 12, 13, 14, 15);
}

// B-matrix 32x16 bf16 tile: lane = N column (lane&15); lanes 0-15 hold K 0-7 / 16-23,
// lanes 16-31 hold K 8-15 / 24-31 (strided gather).
DI v16bf load_b_bf16(const __bf16* base, int ld, int lane) {
  const int n  = lane & 15;
  const int kb = (lane >> 4) * 8;
  v16bf b;
#pragma unroll
  for (int s = 0; s < 16; ++s)
    b[s] = base[(size_t)(kb + (s & 7) + (s >> 3) * 16) * ld + n];
  return b;
}

DI v8f wmma_bf16(v16bf a, v16bf b, v8f c) {
  return __builtin_amdgcn_wmma_f32_16x16x32_bf16(false, a, false, b,
                                                 (short)0, c, false, false);
}

// C/D layout: VGPR j -> M = j + 8*(lane>=16), N = lane&15.
DI void store_tile_f32(float* base, size_t ld, int lane, v8f acc, float scl) {
  const int n  = lane & 15;
  const int mo = (lane >> 4) * 8;
#pragma unroll
  for (int j = 0; j < 8; ++j)
    base[(size_t)(mo + j) * ld + n] = acc[j] * scl;
}

struct WPtrs { const float* w[6]; };

// ---------- 1) weights: f32 [D,D] -> bf16 transposed; layout per Q/K/V:
// slot 3p+0 = Wr^T, 3p+1 = Wi^T, 3p+2 = -Wi^T (lets the real part accumulate
// as a single WMMA chain with no negate/subtract fixup). ----------
__global__ void wt_kernel(WPtrs wp, __bf16* wt) {
  const int n = threadIdx.x;   // 0..255
  const int k = blockIdx.x;    // 0..255
  const int w = blockIdx.y;    // 0..5 (pair p = w>>1, ri = w&1)
  const int p  = w >> 1;
  const int ri = w & 1;
  const float val = wp.w[w][(size_t)n * D_ + k];
  wt[(size_t)(3 * p + ri) * (D_ * D_) + (size_t)k * D_ + n] = f2bf(val);
  if (ri)
    wt[(size_t)(3 * p + 2) * (D_ * D_) + (size_t)k * D_ + n] = f2bf(-val);
}

// ---------- 2) complex projections (Q/K/V) via bf16 WMMA ----------
// grid: (B*S/16, D/64, 3); block: 32. One wave = 16x64 strip (4 n-tiles),
// 2 accumulators per n-tile (re, im), A converted once and reused 4x.
// 16 WMMAs per K-step of 32, K = 256.
__global__ void proj_kernel(const float* __restrict__ xr, const float* __restrict__ xi,
                            const __bf16* __restrict__ wt,
                            __bf16* __restrict__ qf, __bf16* __restrict__ kfT,
                            __bf16* __restrict__ vr, __bf16* __restrict__ vi) {
  const int lane = threadIdx.x;
  const int r0 = blockIdx.x * 16;     // row in flattened [B*S)
  const int n0 = blockIdx.y * 64;     // output column strip within D
  const int w  = blockIdx.z;          // 0=Q, 1=K, 2=V
  const __bf16* WTr  = wt + (size_t)(3 * w)     * (D_ * D_);
  const __bf16* WTi  = wt + (size_t)(3 * w + 1) * (D_ * D_);
  const __bf16* WTin = wt + (size_t)(3 * w + 2) * (D_ * D_);
  const float* xr0 = xr + (size_t)r0 * D_;
  const float* xi0 = xi + (size_t)r0 * D_;

  v8f re[4], im[4];
#pragma unroll
  for (int t = 0; t < 4; ++t) { re[t] = zero8(); im[t] = zero8(); }

#pragma clang loop unroll(disable)
  for (int k0 = 0; k0 < D_; k0 += 32) {
    v16bf Ar = load_a_f32(xr0 + k0, D_, lane);
    v16bf Ai = load_a_f32(xi0 + k0, D_, lane);
#pragma unroll
    for (int t = 0; t < 4; ++t) {
      const size_t off = (size_t)k0 * D_ + n0 + 16 * t;
      v16bf Br  = load_b_bf16(WTr  + off, D_, lane);
      v16bf Bi  = load_b_bf16(WTi  + off, D_, lane);
      v16bf Bin = load_b_bf16(WTin + off, D_, lane);
      re[t] = wmma_bf16(Ar, Br,  re[t]);   //  xr*Wr
      re[t] = wmma_bf16(Ai, Bin, re[t]);   // -xi*Wi
      im[t] = wmma_bf16(Ar, Bi,  im[t]);   //  xr*Wi
      im[t] = wmma_bf16(Ai, Br,  im[t]);   //  xi*Wr
    }
  }

  const int n  = lane & 15;
  const int mo = (lane >> 4) * 8;
  const int b  = r0 / S_;
  const int s0 = r0 % S_;

  if (w == 0) {                       // q_flat [B,S,2D], real | imag halves
    __bf16* q0 = qf + ((size_t)b * S_ + s0) * D2_;
#pragma unroll
    for (int t = 0; t < 4; ++t)
#pragma unroll
      for (int j = 0; j < 8; ++j) {
        q0[(size_t)(mo + j) * D2_ +       n0 + 16 * t + n] = f2bf(re[t][j]);
        q0[(size_t)(mo + j) * D2_ + D_ +  n0 + 16 * t + n] = f2bf(im[t][j]);
      }
  } else if (w == 1) {                // k_flat transposed: kfT [B,2D,S]
    __bf16* kb = kfT + (size_t)b * D2_ * S_;
#pragma unroll
    for (int t = 0; t < 4; ++t)
#pragma unroll
      for (int j = 0; j < 8; ++j) {
        kb[(size_t)(n0 + 16 * t + n)      * S_ + s0 + mo + j] = f2bf(re[t][j]);
        kb[(size_t)(D_ + n0 + 16 * t + n) * S_ + s0 + mo + j] = f2bf(im[t][j]);
      }
  } else {                            // v_r / v_i bf16 [B,S,D]
    __bf16* vr0 = vr + ((size_t)b * S_ + s0) * D_;
    __bf16* vi0 = vi + ((size_t)b * S_ + s0) * D_;
#pragma unroll
    for (int t = 0; t < 4; ++t)
#pragma unroll
      for (int j = 0; j < 8; ++j) {
        vr0[(size_t)(mo + j) * D_ + n0 + 16 * t + n] = f2bf(re[t][j]);
        vi0[(size_t)(mo + j) * D_ + n0 + 16 * t + n] = f2bf(im[t][j]);
      }
  }
}

// ---------- 3) scores = scale * q_flat @ k_flat^T  (raw, pre-softmax) ----------
// grid: (S/32, S/64, B); block 32. Each wave: 32x64 output (2 A-tiles x
// 4 B-tiles -> 8 WMMAs per K-step); K = 512 -> 16 K-steps.
__global__ void scores_kernel(const __bf16* __restrict__ qf,
                              const __bf16* __restrict__ kfT,
                              float* __restrict__ attn) {
  const int lane = threadIdx.x;
  const int q0 = blockIdx.x * 32;
  const int c0 = blockIdx.y * 64;
  const int b  = blockIdx.z;
  const __bf16* qrow0 = qf  + ((size_t)b * S_ + q0) * D2_;
  const __bf16* qrow1 = qrow0 + (size_t)16 * D2_;
  const __bf16* kbase = kfT + (size_t)b * D2_ * S_;

  v8f acc[2][4];
#pragma unroll
  for (int r = 0; r < 2; ++r)
#pragma unroll
    for (int t = 0; t < 4; ++t) acc[r][t] = zero8();

#pragma clang loop unroll(disable)
  for (int d0 = 0; d0 < D2_; d0 += 32) {
    v16bf A0 = load_a_bf16(qrow0 + d0, D2_, lane);
    v16bf A1 = load_a_bf16(qrow1 + d0, D2_, lane);
#pragma unroll
    for (int t = 0; t < 4; ++t) {
      v16bf Bt = load_b_bf16(kbase + (size_t)d0 * S_ + c0 + 16 * t, S_, lane);
      acc[0][t] = wmma_bf16(A0, Bt, acc[0][t]);
      acc[1][t] = wmma_bf16(A1, Bt, acc[1][t]);
    }
  }
  const float scale = 0.0625f;   // 256^-0.5
#pragma unroll
  for (int r = 0; r < 2; ++r) {
    float* ob = attn + ((size_t)b * S_ + q0 + 16 * r) * S_ + c0;
#pragma unroll
    for (int t = 0; t < 4; ++t)
      store_tile_f32(ob + 16 * t, S_, lane, acc[r][t], scale);
  }
}

// ---------- 4) row softmax in place over attn [B*S rows of S] ----------
// One 256-thread block per row; row held in registers (16 f32/thread),
// LDS tree reductions for max and sum.
__global__ void softmax_kernel(float* __restrict__ attn) {
  __shared__ float red[256];
  const int tid = threadIdx.x;
  float* row = attn + (size_t)blockIdx.x * S_;

  float v[16];
  float mx = -__builtin_inff();
#pragma unroll
  for (int i = 0; i < 16; ++i) { v[i] = row[tid + i * 256]; mx = fmaxf(mx, v[i]); }
  red[tid] = mx;
  __syncthreads();
  for (int off = 128; off > 0; off >>= 1) {
    if (tid < off) red[tid] = fmaxf(red[tid], red[tid + off]);
    __syncthreads();
  }
  mx = red[0];
  __syncthreads();

  float s = 0.f;
#pragma unroll
  for (int i = 0; i < 16; ++i) { v[i] = __expf(v[i] - mx); s += v[i]; }
  red[tid] = s;
  __syncthreads();
  for (int off = 128; off > 0; off >>= 1) {
    if (tid < off) red[tid] += red[tid + off];
    __syncthreads();
  }
  const float inv = 1.0f / red[0];
#pragma unroll
  for (int i = 0; i < 16; ++i) row[tid + i * 256] = v[i] * inv;
}

// ---------- 5) out_r = attn @ v_r, out_i = attn @ v_i ----------
// grid: (S/16, B); block 256 (8 waves). One block owns one 16-row q-tile and
// produces ALL 256 output columns for both real and imag: wave w covers
// n in [32w, 32w+32) with 4 accumulators (2 n-tiles x {r,i}).
// attn is read ONCE from HBM per q-tile (8 waves' A reads share WGP$/L2).
__global__ void out_kernel(const float* __restrict__ attn,
                           const __bf16* __restrict__ vr, const __bf16* __restrict__ vi,
                           float* __restrict__ outr, float* __restrict__ outi) {
  const int lane = threadIdx.x & 31;
  const int wave = threadIdx.x >> 5;      // 0..7
  const int q0 = blockIdx.x * 16;
  const int b  = blockIdx.y;
  const int n0 = wave * 32;               // two 16-wide n tiles: n0, n0+16
  const float*  arow = attn + ((size_t)b * S_ + q0) * S_;
  const __bf16* vrb  = vr + (size_t)b * S_ * D_;
  const __bf16* vib  = vi + (size_t)b * S_ * D_;

  v8f ar0 = zero8(), ar1 = zero8(), ai0 = zero8(), ai1 = zero8();
#pragma clang loop unroll(disable)
  for (int k0 = 0; k0 < S_; k0 += 32) {
    __builtin_prefetch(arow + k0 + 32, 0, 0);
    v16bf A   = load_a_f32(arow + k0, S_, lane);
    v16bf Br0 = load_b_bf16(vrb + (size_t)k0 * D_ + n0,      D_, lane);
    v16bf Br1 = load_b_bf16(vrb + (size_t)k0 * D_ + n0 + 16, D_, lane);
    v16bf Bi0 = load_b_bf16(vib + (size_t)k0 * D_ + n0,      D_, lane);
    v16bf Bi1 = load_b_bf16(vib + (size_t)k0 * D_ + n0 + 16, D_, lane);
    ar0 = wmma_bf16(A, Br0, ar0);
    ar1 = wmma_bf16(A, Br1, ar1);
    ai0 = wmma_bf16(A, Bi0, ai0);
    ai1 = wmma_bf16(A, Bi1, ai1);
  }
  float* orow = outr + ((size_t)b * S_ + q0) * D_ + n0;
  float* irow = outi + ((size_t)b * S_ + q0) * D_ + n0;
  store_tile_f32(orow,      D_, lane, ar0, 1.0f);
  store_tile_f32(orow + 16, D_, lane, ar1, 1.0f);
  store_tile_f32(irow,      D_, lane, ai0, 1.0f);
  store_tile_f32(irow + 16, D_, lane, ai1, 1.0f);
}

// ---------- launch ----------
extern "C" void kernel_launch(void* const* d_in, const int* in_sizes, int n_in,
                              void* d_out, int out_size, void* d_ws, size_t ws_size,
                              hipStream_t stream) {
  (void)in_sizes; (void)n_in; (void)out_size; (void)ws_size;

  const float* xr = (const float*)d_in[0];
  const float* xi = (const float*)d_in[1];
  WPtrs wp;
  for (int i = 0; i < 6; ++i) wp.w[i] = (const float*)d_in[2 + i];

  // workspace layout (bytes)
  char* ws = (char*)d_ws;
  const size_t WT_B  = (size_t)9 * D_ * D_ * 2;            //  1.125 MB
  const size_t QF_B  = (size_t)B_ * S_ * D2_ * 2;          //   16 MB
  const size_t KFT_B = QF_B;                               //   16 MB
  const size_t VR_B  = (size_t)B_ * S_ * D_ * 2;           //    8 MB
  __bf16* wt  = (__bf16*)(ws);
  __bf16* qf  = (__bf16*)(ws + WT_B);
  __bf16* kfT = (__bf16*)(ws + WT_B + QF_B);
  __bf16* vr  = (__bf16*)(ws + WT_B + QF_B + KFT_B);
  __bf16* vi  = (__bf16*)(ws + WT_B + QF_B + KFT_B + VR_B);

  float* outr = (float*)d_out;
  float* outi = outr + (size_t)B_ * S_ * D_;
  float* attn = outr + (size_t)2 * B_ * S_ * D_;

  wt_kernel<<<dim3(D_, 6), D_, 0, stream>>>(wp, wt);
  proj_kernel<<<dim3((B_ * S_) / 16, D_ / 64, 3), 32, 0, stream>>>(
      xr, xi, wt, qf, kfT, vr, vi);
  scores_kernel<<<dim3(S_ / 32, S_ / 64, B_), 32, 0, stream>>>(qf, kfT, attn);
  softmax_kernel<<<dim3(B_ * S_), 256, 0, stream>>>(attn);
  out_kernel<<<dim3(S_ / 16, B_), 256, 0, stream>>>(attn, vr, vi, outr, outi);
}
